// NonLocalMeans_14199161880784
// MI455X (gfx1250) — compile-verified, matrix operand order checked by reading, not run on hardware
//
#include <hip/hip_runtime.h>
#include <hip/hip_bf16.h>
#include <math.h>

#define N_B   4
#define C_IN  256
#define CH    128
#define HW    4096
#define QK_SCALE (1.0f/16.0f)   // 1/sqrt(256), folded into theta weights

typedef __bf16 bf16_t;
typedef __attribute__((ext_vector_type(16))) __bf16 v16bf;
typedef __attribute__((ext_vector_type(8)))  float  v8f;

__device__ __forceinline__ v8f wmma_bf16(v16bf a, v16bf b, v8f c) {
  // 8 args: (neg_a, A, neg_b, B, c_mod, C, reuse_a, reuse_b)
  return __builtin_amdgcn_wmma_f32_16x16x32_bf16(false, a, false, b, (short)0, c, false, false);
}

// gfx1250 async copy global -> LDS (16B per lane), tracked by ASYNCcnt.
// The low 32 bits of a generic pointer into LDS are the LDS byte offset.
__device__ __forceinline__ void async_copy_b128(void* lds_ptr, const void* gptr) {
  unsigned int lds_addr = (unsigned int)(size_t)lds_ptr;
  asm volatile("global_load_async_to_lds_b128 %0, %1, off"
               :: "v"(lds_addr), "v"(gptr) : "memory");
}
__device__ __forceinline__ void wait_async0() {
  asm volatile("s_wait_asynccnt 0x0" ::: "memory");
}

// ---------------------------------------------------------------------------
// Kernel 0a: pack theta_w||phi_w -> bf16 [256][256], theta_b||phi_b -> f32[256]
// theta half pre-scaled by 1/sqrt(C) so attention scores need no extra scale.
// ---------------------------------------------------------------------------
__global__ void k_convert_w(const float* __restrict__ theta_w, const float* __restrict__ theta_b,
                            const float* __restrict__ phi_w,   const float* __restrict__ phi_b,
                            bf16_t* __restrict__ w_bf, float* __restrict__ bias) {
  int dd = blockIdx.x;          // 0..255 (0..127 = theta*scale, 128..255 = phi)
  int c  = threadIdx.x;         // 0..255
  const bool is_theta = dd < CH;
  const float* src = is_theta ? (theta_w + (size_t)dd * C_IN) : (phi_w + (size_t)(dd - CH) * C_IN);
  float scl = is_theta ? QK_SCALE : 1.0f;
  w_bf[(size_t)dd * C_IN + c] = (bf16_t)(src[c] * scl);
  if (c == 0) bias[dd] = (is_theta ? theta_b[dd] : phi_b[dd - CH]) * scl;
}

// ---------------------------------------------------------------------------
// Kernel 0b: x(f32,[n][c][hw]) -> x_bf (same layout, V operand)
//                              -> xT_bf ([n][hw][c], A operand for projection)
// ---------------------------------------------------------------------------
__global__ void k_transpose(const float* __restrict__ x,
                            bf16_t* __restrict__ x_bf, bf16_t* __restrict__ xT_bf) {
  __shared__ bf16_t tile[32][33];
  const int n  = blockIdx.z;
  const int c0 = blockIdx.y * 32;
  const int s0 = blockIdx.x * 32;
  const int tx = threadIdx.x, ty = threadIdx.y;
  #pragma unroll
  for (int k = 0; k < 4; ++k) {
    int row = ty + 8 * k;
    size_t idx = ((size_t)(n * C_IN + c0 + row)) * HW + s0 + tx;
    bf16_t b = (bf16_t)x[idx];
    x_bf[idx] = b;
    tile[row][tx] = b;
  }
  __syncthreads();
  #pragma unroll
  for (int k = 0; k < 4; ++k) {
    int row = ty + 8 * k;
    xT_bf[((size_t)(n * HW + s0 + row)) * C_IN + c0 + tx] = tile[tx][row];
  }
}

// ---------------------------------------------------------------------------
// Kernel 1: projection GEMM  qk[n][s][dd] = sum_c xT[n][s][c] * w[dd][c] + b[dd]
// ---------------------------------------------------------------------------
__global__ void k_proj(const bf16_t* __restrict__ xT_bf, const bf16_t* __restrict__ w_bf,
                       const float* __restrict__ bias, bf16_t* __restrict__ qk_bf) {
  const int lane = threadIdx.x & 31;
  const int wave = threadIdx.x >> 5;
  const int n = blockIdx.y;
  const int s_tile = blockIdx.x;
  const int dd_base = wave * 64;
  const int lo = lane & 15, hi = lane >> 4;

  const bf16_t* aRow = xT_bf + ((size_t)(n * HW + s_tile * 16 + lo)) * C_IN + hi * 16;

  v8f acc[4];
  #pragma unroll
  for (int j = 0; j < 4; ++j) {
    float b = bias[dd_base + j * 16 + lo];
    #pragma unroll
    for (int r = 0; r < 8; ++r) acc[j][r] = b;
  }
  #pragma unroll
  for (int kc = 0; kc < 8; ++kc) {
    v16bf a = *(const v16bf*)(aRow + kc * 32);
    #pragma unroll
    for (int j = 0; j < 4; ++j) {
      const bf16_t* bRow = w_bf + (size_t)(dd_base + j * 16 + lo) * C_IN + kc * 32 + hi * 16;
      v16bf bb = *(const v16bf*)bRow;
      acc[j] = wmma_bf16(a, bb, acc[j]);
    }
  }
  #pragma unroll
  for (int j = 0; j < 4; ++j) {
    int dd = dd_base + j * 16 + lo;
    #pragma unroll
    for (int r = 0; r < 8; ++r) {
      int s = s_tile * 16 + hi * 8 + r;
      qk_bf[((size_t)(n * HW + s)) * 256 + dd] = (bf16_t)acc[j][r];
    }
  }
}

// ---------------------------------------------------------------------------
// Kernel 2: flash attention with double-buffered async K/V staging in LDS.
// Per wave: 16 query rows, full 256-ch output tile in 16 v8f accumulators.
// grid (HW/128, N), block 256 (8 waves)
// ---------------------------------------------------------------------------
#define ATTN_WAVES 8
#define KPAD 144   // 128 + 16: 288B rows, 32B-aligned, staggers LDS banks
#define VPAD 48    // 32 + 16:   96B rows, 32B-aligned
__global__ void __launch_bounds__(256) k_attn(const bf16_t* __restrict__ qk_bf,
                                              const bf16_t* __restrict__ x_bf,
                                              float* __restrict__ out) {
  __shared__ bf16_t kx[2][32][KPAD];        // K tile: [t][d]
  __shared__ bf16_t vx[2][256][VPAD];       // V tile: [i][t]
  __shared__ bf16_t p_lds[ATTN_WAVES][16][VPAD];
  const int tid  = threadIdx.x;
  const int lane = tid & 31;
  const int wave = tid >> 5;
  const int n = blockIdx.y;
  const int s_base = blockIdx.x * (16 * ATTN_WAVES) + wave * 16;
  const int lo = lane & 15, hi = lane >> 4;

  const bf16_t* kBase = qk_bf + ((size_t)n * HW) * 256 + 128;  // k half
  const bf16_t* vBase = x_bf + ((size_t)n * C_IN) * HW;        // V[t][i] = x[i][t]

  // Q fragments (A layout), loaded once from global
  v16bf qfrag[4];
  {
    const bf16_t* qRow = qk_bf + ((size_t)(n * HW + s_base + lo)) * 256 + hi * 16;
    #pragma unroll
    for (int kc = 0; kc < 4; ++kc) qfrag[kc] = *(const v16bf*)(qRow + kc * 32);
  }

  v8f acc[16];
  #pragma unroll
  for (int it = 0; it < 16; ++it) {
    #pragma unroll
    for (int r = 0; r < 8; ++r) acc[it][r] = 0.0f;
  }
  float m_run[8], l_run[8];
  #pragma unroll
  for (int r = 0; r < 8; ++r) { m_run[r] = -INFINITY; l_run[r] = 0.0f; }

  // cooperative async stage of one t-block (32 t): K 32x128, V 256x32 (16B/lane chunks)
  auto stage = [&](int tblk, int buf) {
    #pragma unroll
    for (int kk = 0; kk < 2; ++kk) {            // K: 512 chunks / 256 threads
      int c = tid + kk * 256;
      int t = c >> 4, doff = (c & 15) * 8;
      async_copy_b128(&kx[buf][t][doff], kBase + ((size_t)(tblk + t)) * 256 + doff);
    }
    #pragma unroll
    for (int kk = 0; kk < 4; ++kk) {            // V: 1024 chunks / 256 threads
      int c = tid + kk * 256;
      int i = c >> 2, toff = (c & 3) * 8;
      async_copy_b128(&vx[buf][i][toff], vBase + (size_t)i * HW + tblk + toff);
    }
  };

  stage(0, 0);
  const int NT = HW / 32;
  for (int itb = 0; itb < NT; ++itb) {
    const int cur = itb & 1;
    wait_async0();              // own cur-buffer copies done (next not yet issued)
    __syncthreads();            // cur visible to all; all done reading buffer cur^1
    if (itb + 1 < NT) stage((itb + 1) * 32, cur ^ 1);

    // ---- scores S[s][t] = sum_d q[s][d] k[t][d]  (scale folded into q)
    v8f S[2];
    #pragma unroll
    for (int tt = 0; tt < 2; ++tt) {
      v8f s0 = {};
      #pragma unroll
      for (int kc = 0; kc < 4; ++kc) {
        v16bf kf = *(const v16bf*)&kx[cur][tt * 16 + lo][kc * 32 + hi * 16];
        s0 = wmma_bf16(qfrag[kc], kf, s0);
      }
      S[tt] = s0;
    }

    // ---- online softmax: row stats per (lane,r), row = hi*8 + r
    float mnew[8];
    #pragma unroll
    for (int r = 0; r < 8; ++r) {
      float mx = fmaxf(S[0][r], S[1][r]);
      #pragma unroll
      for (int off = 8; off >= 1; off >>= 1)
        mx = fmaxf(mx, __shfl_xor(mx, off, 32));
      mnew[r] = fmaxf(m_run[r], mx);
    }
    float alpha[8];
    #pragma unroll
    for (int r = 0; r < 8; ++r) {
      alpha[r] = __expf(m_run[r] - mnew[r]);
      m_run[r] = mnew[r];
      l_run[r] *= alpha[r];
    }
    #pragma unroll
    for (int it = 0; it < 16; ++it) {
      #pragma unroll
      for (int r = 0; r < 8; ++r) acc[it][r] *= alpha[r];
    }
    #pragma unroll
    for (int r = 0; r < 8; ++r) {
      float p0 = __expf(S[0][r] - m_run[r]);
      float p1 = __expf(S[1][r] - m_run[r]);
      float rs = p0 + p1;
      #pragma unroll
      for (int off = 8; off >= 1; off >>= 1)
        rs += __shfl_xor(rs, off, 32);
      l_run[r] += rs;
      p_lds[wave][hi * 8 + r][lo]      = (bf16_t)p0;
      p_lds[wave][hi * 8 + r][16 + lo] = (bf16_t)p1;
    }

    // re-read P in A layout (wave-local LDS: same-wave DS ops are in order)
    v16bf pfrag = *(const v16bf*)&p_lds[wave][lo][hi * 16];

    // ---- out[s][i] += P[s][t] * V[t][i]
    #pragma unroll
    for (int it = 0; it < 16; ++it) {
      v16bf vf = *(const v16bf*)&vx[cur][it * 16 + lo][hi * 16];
      acc[it] = wmma_bf16(pfrag, vf, acc[it]);
    }
    __syncthreads();            // everyone done reading cur before it is refilled
  }

  // ---- finalize: divide by l, store out[n][i][s] (r-consecutive => s-contiguous)
  float inv_l[8];
  #pragma unroll
  for (int r = 0; r < 8; ++r) inv_l[r] = 1.0f / l_run[r];
  #pragma unroll
  for (int it = 0; it < 16; ++it) {
    int i = it * 16 + lo;
    float* o = out + ((size_t)(n * C_IN + i)) * HW + s_base + hi * 8;
    float4 v0 = make_float4(acc[it][0] * inv_l[0], acc[it][1] * inv_l[1],
                            acc[it][2] * inv_l[2], acc[it][3] * inv_l[3]);
    float4 v1 = make_float4(acc[it][4] * inv_l[4], acc[it][5] * inv_l[5],
                            acc[it][6] * inv_l[6], acc[it][7] * inv_l[7]);
    *(float4*)(o)     = v0;
    *(float4*)(o + 4) = v1;
  }
}

// ---------------------------------------------------------------------------
extern "C" void kernel_launch(void* const* d_in, const int* in_sizes, int n_in,
                              void* d_out, int out_size, void* d_ws, size_t ws_size,
                              hipStream_t stream) {
  const float* x       = (const float*)d_in[0];
  const float* theta_w = (const float*)d_in[1];
  const float* theta_b = (const float*)d_in[2];
  const float* phi_w   = (const float*)d_in[3];
  const float* phi_b   = (const float*)d_in[4];
  float* out = (float*)d_out;

  char* ws = (char*)d_ws;
  const size_t xbf_bytes = (size_t)N_B * C_IN * HW * sizeof(bf16_t);   // 8 MB
  bf16_t* x_bf  = (bf16_t*)(ws);
  bf16_t* xT_bf = (bf16_t*)(ws + xbf_bytes);
  bf16_t* w_bf  = (bf16_t*)(ws + 2 * xbf_bytes);
  float*  bias  = (float*) (ws + 2 * xbf_bytes + 256 * 256 * sizeof(bf16_t));
  bf16_t* qk_bf = (bf16_t*)(ws + 2 * xbf_bytes + 256 * 256 * sizeof(bf16_t) + 4096);

  k_convert_w<<<dim3(256), dim3(256), 0, stream>>>(theta_w, theta_b, phi_w, phi_b, w_bf, bias);
  k_transpose<<<dim3(HW / 32, C_IN / 32, N_B), dim3(32, 8), 0, stream>>>(x, x_bf, xT_bf);
  k_proj<<<dim3(HW / 16, N_B), dim3(128), 0, stream>>>(xT_bf, w_bf, bias, qk_bf);
  k_attn<<<dim3(HW / (16 * ATTN_WAVES), N_B), dim3(256), 0, stream>>>(qk_bf, x_bf, out);
}
